// KeyValAttention_1726576856831
// MI455X (gfx1250) — matrix operand order repaired and checked by reading, not compiled, not run on hardware
//
#include <hip/hip_runtime.h>
#include <hip/hip_bf16.h>

typedef __attribute__((ext_vector_type(16))) __bf16 v16bf;
typedef __attribute__((ext_vector_type(8)))  float  v8f;
typedef __attribute__((ext_vector_type(4)))  float  v4f;
typedef __attribute__((ext_vector_type(4)))  int    v4i;

namespace {
constexpr int B_ = 2, H_ = 16, J_ = 2048, K_ = 2048, M_ = 64;
constexpr int TJ = 64;            // J rows per workgroup
constexpr int CK = 128;           // K columns per streamed chunk
constexpr int NCHUNK = K_ / CK;   // 16
constexpr int NTHREADS = 256;     // 8 wave32
}

static __device__ __forceinline__ unsigned short f2bfbits(float f) {
  union { float f; unsigned u; } x; x.f = f;
  unsigned r = x.u + 0x7FFFu + ((x.u >> 16) & 1u);   // round-to-nearest-even
  return (unsigned short)(r >> 16);
}
static __device__ __forceinline__ __bf16 bfbits(unsigned short b) {
  __bf16 h; __builtin_memcpy(&h, &b, sizeof(h)); return h;
}
static __device__ __forceinline__ unsigned long long pack4bf(float a, float b,
                                                             float c, float d) {
  return (unsigned long long)f2bfbits(a)
       | ((unsigned long long)f2bfbits(b) << 16)
       | ((unsigned long long)f2bfbits(c) << 32)
       | ((unsigned long long)f2bfbits(d) << 48);
}

__global__ __launch_bounds__(NTHREADS)
void attn_fused(const float* __restrict__ qg,
                const float* __restrict__ kg,
                const float* __restrict__ vg,
                const int*   __restrict__ maskg,
                const float* __restrict__ amg,
                float* __restrict__ outg,
                float* __restrict__ attg)
{
  __shared__ __align__(16) unsigned short sQ[TJ][M_];    // bf16, pre-scaled Q
  __shared__ __align__(16) unsigned short sKV[CK][M_];   // bf16, K rows / V rows
  __shared__ __align__(16) unsigned short sA[TJ][CK];    // bf16, finalized attn chunk
  __shared__ float sRed[NTHREADS];
  __shared__ float sMax[TJ];
  __shared__ float sInv[TJ];

  const int tid  = threadIdx.x;
  const int lane = tid & 31;
  const int wv   = tid >> 5;               // wave id 0..7 (uniform per wave)

  const int jtiles = J_ / TJ;              // 32
  const int jt_id  = blockIdx.x % jtiles;
  const int bh     = blockIdx.x / jtiles;  // 0..31
  const int j0     = jt_id * TJ;

  const float* qb = qg + (size_t)bh * J_ * M_;
  const float* kb = kg + (size_t)bh * K_ * M_;
  const float* vb = vg + (size_t)bh * K_ * M_;
  float* ob = outg + (size_t)bh * J_ * M_;
  float* ab = attg + (size_t)bh * J_ * K_;

  const v4f* qb4 = (const v4f*)qb;
  const v4f* kb4 = (const v4f*)kb;
  const v4f* vb4 = (const v4f*)vb;

  // ---------- load + scale Q tile (1/sqrt(64)=0.125) as v4f -> bf16x4 ----------
  for (int idx = tid; idx < TJ * M_ / 4; idx += NTHREADS) {
    v4f v = qb4[(size_t)j0 * (M_ / 4) + idx];
    *(unsigned long long*)&sQ[idx >> 4][(idx & 15) * 4] =
        pack4bf(v.x * 0.125f, v.y * 0.125f, v.z * 0.125f, v.w * 0.125f);
  }
  __syncthreads();

  // ---------- prebuild Q A-fragments (A 16x32 bf16 layout) ----------
  const int jt1 = wv & 3;        // j sub-tile (GEMM1)
  const int ntb = wv >> 2;       // n sub-tiles ntb + s*2, s=0..3 (GEMM1)
  v16bf aq[2];
  #pragma unroll
  for (int ks = 0; ks < 2; ++ks) {
    #pragma unroll
    for (int e = 0; e < 16; ++e) {
      int kc = ks * 32 + (e & 7) + ((e & 8) << 1) + ((lane & 16) >> 1);
      aq[ks][e] = bfbits(sQ[jt1 * 16 + (lane & 15)][kc]);
    }
  }

  // ================= Phase 1: raw scores S = Qs @ K^T -> attn slice ==========
  for (int c = 0; c < NCHUNK; ++c) {
    if (c + 1 < NCHUNK)
      __builtin_prefetch(kb + (size_t)(c + 1) * CK * M_ + (size_t)tid * 32, 0, 0);
    for (int idx = tid; idx < CK * M_ / 4; idx += NTHREADS) {
      v4f v = kb4[(size_t)c * (CK * M_ / 4) + idx];
      *(unsigned long long*)&sKV[idx >> 4][(idx & 15) * 4] =
          pack4bf(v.x, v.y, v.z, v.w);
    }
    __syncthreads();

    #pragma unroll
    for (int s = 0; s < 4; ++s) {
      const int nt = ntb + s * 2;
      v8f acc = {};
      #pragma unroll
      for (int ks = 0; ks < 2; ++ks) {
        v16bf bk;   // B 32x16: k = ks*32 + (e&7) + ((e&8)<<1) + ((lane&16)>>1), n = lane&15
        #pragma unroll
        for (int e = 0; e < 16; ++e)
          bk[e] = bfbits(sKV[nt * 16 + (lane & 15)][ks * 32 + (e & 7) + ((e & 8) << 1) + ((lane & 16) >> 1)]);
        acc = __builtin_amdgcn_wmma_f32_16x16x32_bf16(false, aq[ks], false, bk,
                                                      (short)0, acc, false, false);
      }
      #pragma unroll
      for (int r = 0; r < 8; ++r) {
        int jl = jt1 * 16 + r + ((lane & 16) >> 1);
        int kc = c * CK + nt * 16 + (lane & 15);
        ab[(size_t)(j0 + jl) * K_ + kc] = acc[r];   // keep temporal: L2-resident
      }
    }
    __syncthreads();
  }

  __threadfence();       // make raw-score stores visible block-wide (via L2)
  __syncthreads();

  // ================= Phase 2: softmax row stats over L2-resident S ===========
  {
    const int row = tid >> 2;     // 64 rows, 4 threads each
    const int g   = tid & 3;
    const v4f* srow4 = (const v4f*)(ab + (size_t)(j0 + row) * K_);
    float mx = -3.402823466e38f;
    for (int i = g; i < K_ / 4; i += 4) {
      v4f v = srow4[i];
      mx = fmaxf(mx, fmaxf(fmaxf(v.x, v.y), fmaxf(v.z, v.w)));
    }
    sRed[tid] = mx;
    __syncthreads();
    if (g == 0) {
      float m2 = sRed[tid];
      #pragma unroll
      for (int t = 1; t < 4; ++t) m2 = fmaxf(m2, sRed[tid + t]);
      sMax[row] = m2;
    }
    __syncthreads();
    const float rm = sMax[row];
    float sm = 0.f;
    for (int i = g; i < K_ / 4; i += 4) {
      v4f v = srow4[i];
      sm += __expf(v.x - rm) + __expf(v.y - rm) + __expf(v.z - rm) + __expf(v.w - rm);
    }
    sRed[tid] = sm;
    __syncthreads();
    if (g == 0) {
      float s2 = 0.f;
      #pragma unroll
      for (int t = 0; t < 4; ++t) s2 += sRed[tid + t];
      sInv[row] = 1.0f / s2;
    }
    __syncthreads();
  }

  // ============ Phase 3: finalize attn (post-softmax masks) + out=attn@V =====
  const int jt2 = wv & 3;
  const int mtb = wv >> 2;       // M slabs mtb and mtb+2
  v8f oacc[2] = {{}, {}};

  for (int c = 0; c < NCHUNK; ++c) {
    if (c + 1 < NCHUNK)
      __builtin_prefetch(vb + (size_t)(c + 1) * CK * M_ + (size_t)tid * 32, 0, 0);
    // V chunk -> LDS bf16 (vectorized)
    for (int idx = tid; idx < CK * M_ / 4; idx += NTHREADS) {
      v4f v = vb4[(size_t)c * (CK * M_ / 4) + idx];
      *(unsigned long long*)&sKV[idx >> 4][(idx & 15) * 4] =
          pack4bf(v.x, v.y, v.z, v.w);
    }

    // finalize this attn chunk: exp*invsum, post-softmax where()+additive mask
    {
      const int row = tid >> 2;          // 64 rows, 4 threads each
      const int cb  = (tid & 3) * 32;    // 32 cols per thread, as 8 v4f
      const float rm = sMax[row], ri = sInv[row];
      const size_t rbase = (size_t)(j0 + row) * K_ + (size_t)c * CK;
      #pragma unroll
      for (int t4 = 0; t4 < 8; ++t4) {
        const int col = cb + t4 * 4;
        const size_t gi = rbase + col;   // same flat index works for [J,K] masks
        v4f sv = *(const v4f*)(ab + gi);
        v4i mv = *(const v4i*)(maskg + gi);
        v4f av = *(const v4f*)(amg + gi);
        v4f a;
        a.x = (mv.x == 0 ? 1e-9f : __expf(sv.x - rm) * ri) + av.x;
        a.y = (mv.y == 0 ? 1e-9f : __expf(sv.y - rm) * ri) + av.y;
        a.z = (mv.z == 0 ? 1e-9f : __expf(sv.z - rm) * ri) + av.z;
        a.w = (mv.w == 0 ? 1e-9f : __expf(sv.w - rm) * ri) + av.w;
        __builtin_nontemporal_store(a, (v4f*)(ab + gi));   // final attn, dead data
        *(unsigned long long*)&sA[row][col] = pack4bf(a.x, a.y, a.z, a.w);
      }
    }
    __syncthreads();

    // out += attn_chunk @ V_chunk : contraction CK=128 = 4 WMMA k-steps
    #pragma unroll
    for (int ks = 0; ks < 4; ++ks) {
      v16bf aa;
      #pragma unroll
      for (int e = 0; e < 16; ++e) {
        int kc = ks * 32 + (e & 7) + ((e & 8) << 1) + ((lane & 16) >> 1);
        aa[e] = bfbits(sA[jt2 * 16 + (lane & 15)][kc]);
      }
      #pragma unroll
      for (int u = 0; u < 2; ++u) {
        const int mt = mtb + u * 2;
        v16bf bb;
        #pragma unroll
        for (int e = 0; e < 16; ++e)
          bb[e] = bfbits(sKV[ks * 32 + (e & 7) + ((e & 8) << 1) + ((lane & 16) >> 1)][mt * 16 + (lane & 15)]);
        oacc[u] = __builtin_amdgcn_wmma_f32_16x16x32_bf16(false, aa, false, bb,
                                                          (short)0, oacc[u], false, false);
      }
    }
    __syncthreads();   // protect sKV/sA before next chunk overwrites
  }

  // ================= Phase 4: write out tile (never re-read -> NT) ===========
  #pragma unroll
  for (int u = 0; u < 2; ++u) {
    #pragma unroll
    for (int r = 0; r < 8; ++r) {
      int jl = jt2 * 16 + r + ((lane & 16) >> 1);
      int m  = (mtb + u * 2) * 16 + (lane & 15);
      __builtin_nontemporal_store(oacc[u][r], ob + (size_t)(j0 + jl) * M_ + m);
    }
  }
}

extern "C" void kernel_launch(void* const* d_in, const int* in_sizes, int n_in,
                              void* d_out, int out_size, void* d_ws, size_t ws_size,
                              hipStream_t stream) {
  const float* q  = (const float*)d_in[0];
  const float* k  = (const float*)d_in[1];
  const float* v  = (const float*)d_in[2];
  const int*   mk = (const int*)  d_in[3];
  const float* am = (const float*)d_in[4];

  float* outp  = (float*)d_out;                                // [B,H,J,M]
  float* attnp = outp + (size_t)B_ * H_ * J_ * M_;             // [B,H,J,K]

  dim3 grid(B_ * H_ * (J_ / TJ));   // 1024 workgroups
  attn_fused<<<grid, NTHREADS, 0, stream>>>(q, k, v, mk, am, outp, attnp);

  (void)in_sizes; (void)n_in; (void)out_size; (void)d_ws; (void)ws_size;
}
